// SelfAttentionPooling_5944234737961
// MI455X (gfx1250) — matrix operand order repaired and checked by reading, not compiled, not run on hardware
//
#include <hip/hip_runtime.h>
#include <hip/hip_bf16.h>

#define Nn 8192
#define Hh 256
#define Gg 256
#define NMAX 128
#define DSTR 129  // padded dot row stride (floats) to avoid 64-bank conflicts
#define RB 64     // rows per GEMM block (4 row-tiles of 16 per wave)

typedef __attribute__((ext_vector_type(16))) __bf16   v16bf;
typedef __attribute__((ext_vector_type(4)))  __bf16   v4bf;
typedef __attribute__((ext_vector_type(8)))  float    v8f;
typedef __attribute__((ext_vector_type(4)))  unsigned v4u;

union FragB16 { v16bf v; v4u q[2]; };

// ---------------------------------------------------------------------------
// WT[j][k] = (bf16) W[k][j]   (256x256)
// ---------------------------------------------------------------------------
__global__ void sap_transpose_to_bf16(const float* __restrict__ W, __bf16* __restrict__ WT) {
    int j = blockIdx.x;          // output row (= input column)
    int k = threadIdx.x;         // output col (= input row)
    WT[j * Hh + k] = (__bf16)W[k * Hh + j];
}

// ---------------------------------------------------------------------------
// starts[g] = lower_bound(batch, g); starts[G] = N   (batch is sorted)
// ---------------------------------------------------------------------------
__global__ void sap_seg_starts(const int* __restrict__ batch, int* __restrict__ starts) {
    int g = threadIdx.x;         // 256 threads
    int lo = 0, hi = Nn;
    while (lo < hi) {
        int mid = (lo + hi) >> 1;
        if (batch[mid] < g) lo = mid + 1; else hi = mid;
    }
    starts[g] = lo;
    if (g == 0) starts[Gg] = Nn;
}

// ---------------------------------------------------------------------------
// Y = act(X @ W + b) with W pre-transposed in bf16 (WT[j][k] = W[k][j]).
// Block: 512 threads = 16 waves, 64-row x 256-col stripe.
// Wave w owns columns [16w,16w+16); per K-step it loads ONE B fragment plus
// FOUR distinct A fragments (no register WAR), then fires 4 independent
// WMMAs -> single ds clause + back-to-back matrix ops.
// ---------------------------------------------------------------------------
__global__ void sap_gemm_bf16_wmma(const float* __restrict__ X,
                                   const __bf16* __restrict__ WT,
                                   const float* __restrict__ bias,
                                   float* __restrict__ outF,
                                   __bf16* __restrict__ outB,
                                   int do_relu) {
    __shared__ __bf16 inA[RB * Hh];               // 32 KB bf16 input stripe
    const int row0 = blockIdx.x * RB;
    const int tid  = threadIdx.x;

    // Vectorized stage: float4 -> 4x bf16 (8B) into LDS
    {
        const float* Xb = X + (size_t)row0 * Hh;
        for (int idx = tid; idx < RB * Hh / 4; idx += 512) {
            float4 f = ((const float4*)Xb)[idx];
            v4bf p = { (__bf16)f.x, (__bf16)f.y, (__bf16)f.z, (__bf16)f.w };
            *(v4bf*)(inA + idx * 4) = p;
        }
    }
    __syncthreads();

    const int lane = tid & 31;
    const int wv   = tid >> 5;                    // 0..15
    const int m    = lane & 15;
    const int hi   = lane >> 4;                   // 0 or 1
    const int c0   = wv * 16;

    v8f acc0 = {}, acc1 = {}, acc2 = {}, acc3 = {};
    const __bf16* wrow = WT + (size_t)(c0 + m) * Hh;   // this lane's weight row
#pragma unroll
    for (int kt = 0; kt < Hh / 32; ++kt) {
        // B (32x16 bf16): lane holds column c0+m, K = k0..k0+7 , k0+16..k0+23 (+8 for lane>=16)
        FragB16 b;
        const __bf16* pb = wrow + kt * 32 + hi * 8;
        b.q[0] = *(const v4u*)pb;
        b.q[1] = *(const v4u*)(pb + 16);
        if (kt + 1 < Hh / 32) __builtin_prefetch(pb + 32, 0, 3);  // global_prefetch_b8, near

        // A (16x32 bf16): lane<16 -> K = k0..k0+7 , k0+16..k0+23; lane>=16 -> +8
        // Four DISTINCT fragments so all 8 ds_load_b128 can clause together.
        const __bf16* pa = inA + m * Hh + kt * 32 + hi * 8;
        FragB16 a0, a1, a2, a3;
        a0.q[0] = *(const v4u*)(pa + 0 * 16 * Hh);
        a0.q[1] = *(const v4u*)(pa + 0 * 16 * Hh + 16);
        a1.q[0] = *(const v4u*)(pa + 1 * 16 * Hh);
        a1.q[1] = *(const v4u*)(pa + 1 * 16 * Hh + 16);
        a2.q[0] = *(const v4u*)(pa + 2 * 16 * Hh);
        a2.q[1] = *(const v4u*)(pa + 2 * 16 * Hh + 16);
        a3.q[0] = *(const v4u*)(pa + 3 * 16 * Hh);
        a3.q[1] = *(const v4u*)(pa + 3 * 16 * Hh + 16);
        acc0 = __builtin_amdgcn_wmma_f32_16x16x32_bf16(false, a0.v, false, b.v,
                                                       (short)0, acc0, false, false);
        acc1 = __builtin_amdgcn_wmma_f32_16x16x32_bf16(false, a1.v, false, b.v,
                                                       (short)0, acc1, false, false);
        acc2 = __builtin_amdgcn_wmma_f32_16x16x32_bf16(false, a2.v, false, b.v,
                                                       (short)0, acc2, false, false);
        acc3 = __builtin_amdgcn_wmma_f32_16x16x32_bf16(false, a3.v, false, b.v,
                                                       (short)0, acc3, false, false);
    }

    const int col = c0 + m;
    const float bv = bias[col];
    const int rsub = hi ? 8 : 0;
    v8f* accs[4] = { &acc0, &acc1, &acc2, &acc3 };
    for (int it = 0; it < 4; ++it) {
        const int rbase = row0 + it * 16 + rsub;
        for (int v = 0; v < 8; ++v) {
            float val = (*accs[it])[v] + bv;
            if (do_relu) val = fmaxf(val, 0.0f);
            size_t off = (size_t)(rbase + v) * Hh + col;
            if (outF) outF[off] = val;
            if (outB) outB[off] = (__bf16)val;
        }
    }
}

// ---------------------------------------------------------------------------
// Per-graph block-diagonal attention pooling.
// dot = S_blk @ S_blk^T (WMMA), masked softmax reductions, then
// x_agg[g] = sum_j w_j * h[st+j]   with   w_j = sum_i e[i,j] / denom_i.
// One block per graph, 256 threads = 8 waves. Dynamic LDS.
// ---------------------------------------------------------------------------
__global__ void sap_attn_pool(const __bf16* __restrict__ sB,
                              const float* __restrict__ hF,
                              const int* __restrict__ starts,
                              float* __restrict__ xagg) {
    extern __shared__ char smem[];
    __bf16* sS   = (__bf16*)smem;                                   // NMAX*256 bf16
    float*  dotM = (float*)(smem + NMAX * Hh * 2);                  // NMAX*DSTR f32
    float*  em   = (float*)(smem + NMAX * Hh * 2 + NMAX * DSTR * 4);
    float*  dinv = em + NMAX;
    float*  wj   = dinv + NMAX;

    const int g   = blockIdx.x;
    const int tid = threadIdx.x;
    const int st  = starts[g];
    int n = starts[g + 1] - st;
    if (n > NMAX) n = NMAX;                        // practically unreachable
    if (n <= 0) { xagg[g * Hh + tid] = 0.0f; return; }
    const int nt = (n + 15) & ~15;

    // Stage s rows (zero-pad to the 16-multiple)
    for (int idx = tid; idx < nt * Hh; idx += 256) {
        int r = idx >> 8, c = idx & 255;
        sS[idx] = (r < n) ? sB[(size_t)(st + r) * Hh + c] : (__bf16)0.0f;
    }
    __syncthreads();

    const int T    = nt >> 4;
    const int lane = tid & 31;
    const int wv   = tid >> 5;
    const int m    = lane & 15;
    const int hi   = lane >> 4;

    // dot tiles: D = S * S^T ; B fragment is just another s-row (same K order as A)
    for (int t = wv; t < T * T; t += 8) {
        const int it = t / T, jt = t % T;
        v8f acc = {};
#pragma unroll
        for (int kt = 0; kt < Hh / 32; ++kt) {
            FragB16 a, b;
            const __bf16* pa = sS + (it * 16 + m) * Hh + kt * 32 + hi * 8;
            a.q[0] = *(const v4u*)pa;
            a.q[1] = *(const v4u*)(pa + 16);
            const __bf16* pb = sS + (jt * 16 + m) * Hh + kt * 32 + hi * 8;
            b.q[0] = *(const v4u*)pb;
            b.q[1] = *(const v4u*)(pb + 16);
            acc = __builtin_amdgcn_wmma_f32_16x16x32_bf16(false, a.v, false, b.v,
                                                          (short)0, acc, false, false);
        }
        const int j  = jt * 16 + m;
        const int ib = it * 16 + (hi ? 8 : 0);
        for (int v = 0; v < 8; ++v) dotM[(ib + v) * DSTR + j] = acc[v];
    }
    __syncthreads();

    // Row max + denominator (in-block max always >= 0 via the diagonal,
    // so this matches the reference's max(dot*mask) exactly).
    for (int i = tid; i < n; i += 256) {
        float mx = dotM[i * DSTR];
        for (int j = 1; j < n; ++j) mx = fmaxf(mx, dotM[i * DSTR + j]);
        float d = 0.0f;
        for (int j = 0; j < n; ++j) d += __expf(dotM[i * DSTR + j] - mx);
        em[i] = mx;
        dinv[i] = 1.0f / d;
    }
    __syncthreads();

    // Column weights: w_j = sum_i exp(dot[i,j]-m_i) / denom_i
    for (int j = tid; j < n; j += 256) {
        float w = 0.0f;
        for (int i = 0; i < n; ++i) w += __expf(dotM[i * DSTR + j] - em[i]) * dinv[i];
        wj[j] = w;
    }
    __syncthreads();

    // x_agg[g,c] = sum_j w_j * h[st+j, c]   (thread == channel, coalesced)
    const int c = tid;
    float accv = 0.0f;
    for (int j = 0; j < n; ++j) accv += wj[j] * hF[(size_t)(st + j) * Hh + c];
    xagg[g * Hh + c] = accv;
}

// ---------------------------------------------------------------------------
extern "C" void kernel_launch(void* const* d_in, const int* in_sizes, int n_in,
                              void* d_out, int out_size, void* d_ws, size_t ws_size,
                              hipStream_t stream) {
    (void)in_sizes; (void)n_in; (void)out_size; (void)ws_size;
    const float* x    = (const float*)d_in[0];
    const int*   batch= (const int*)d_in[1];
    const float* aw   = (const float*)d_in[3];
    const float* ab   = (const float*)d_in[4];
    const float* p1w  = (const float*)d_in[5];
    const float* p1b  = (const float*)d_in[6];
    const float* p2w  = (const float*)d_in[7];
    const float* p2b  = (const float*)d_in[8];
    const float* r1w  = (const float*)d_in[9];
    const float* r1b  = (const float*)d_in[10];
    const float* r2w  = (const float*)d_in[11];
    const float* r2b  = (const float*)d_in[12];
    float* out = (float*)d_out;

    char* ws = (char*)d_ws;
    float*   t1   = (float*)(ws + 0);                         // 8 MB
    float*   h    = (float*)(ws + (size_t)8388608);           // 8 MB
    __bf16*  sB   = (__bf16*)(ws + (size_t)16777216);         // 4 MB
    float*   xagg = (float*)(ws + (size_t)20971520);          // 256 KB
    float*   t2   = (float*)(ws + (size_t)21233664);          // 256 KB
    __bf16*  WTp1 = (__bf16*)(ws + (size_t)21495808);
    __bf16*  WTp2 = WTp1 + Hh * Hh;
    __bf16*  WTa  = WTp2 + Hh * Hh;
    __bf16*  WTr1 = WTa  + Hh * Hh;
    __bf16*  WTr2 = WTr1 + Hh * Hh;
    int*     starts = (int*)(ws + (size_t)21495808 + (size_t)5 * Hh * Hh * 2);

    sap_transpose_to_bf16<<<Hh, Hh, 0, stream>>>(p1w, WTp1);
    sap_transpose_to_bf16<<<Hh, Hh, 0, stream>>>(p2w, WTp2);
    sap_transpose_to_bf16<<<Hh, Hh, 0, stream>>>(aw,  WTa);
    sap_transpose_to_bf16<<<Hh, Hh, 0, stream>>>(r1w, WTr1);
    sap_transpose_to_bf16<<<Hh, Hh, 0, stream>>>(r2w, WTr2);
    sap_seg_starts<<<1, 256, 0, stream>>>(batch, starts);

    // phi MLP + scores
    sap_gemm_bf16_wmma<<<Nn / RB, 512, 0, stream>>>(x,  WTp1, p1b, t1, nullptr, 1);
    sap_gemm_bf16_wmma<<<Nn / RB, 512, 0, stream>>>(t1, WTp2, p2b, h,  nullptr, 1);
    sap_gemm_bf16_wmma<<<Nn / RB, 512, 0, stream>>>(h,  WTa,  ab, nullptr, sB, 0);

    // block-diagonal attention pooling -> x_agg
    const size_t smem = (size_t)NMAX * Hh * 2     // sS
                      + (size_t)NMAX * DSTR * 4   // dotM (padded stride)
                      + (size_t)3 * NMAX * 4;     // em, dinv, wj
    sap_attn_pool<<<Gg, 256, smem, stream>>>(sB, h, starts, xagg);

    // rho MLP
    sap_gemm_bf16_wmma<<<Gg / RB, 512, 0, stream>>>(xagg, WTr1, r1b, t2,  nullptr, 1);
    sap_gemm_bf16_wmma<<<Gg / RB, 512, 0, stream>>>(t2,   WTr2, r2b, out, nullptr, 1);
}
